// ViM_MambaBlock_86569360818675
// MI455X (gfx1250) — compile-verified, matrix operand order checked by reading, not run on hardware
//
#include <hip/hip_runtime.h>
#include <hip/hip_bf16.h>
#include <math.h>

// Problem constants (B,S,D,E,N,R) = (2,1024,1024,2048,16,64)
#define B_  2
#define S_  1024
#define D_  1024
#define E_  2048
#define N_  16
#define R_  64
#define BS_ (B_*S_)

typedef __attribute__((ext_vector_type(16))) __bf16 v16bf;
typedef __attribute__((ext_vector_type(8)))  float  v8f;
typedef unsigned int u32x4 __attribute__((ext_vector_type(4)));
typedef int          i32x4 __attribute__((ext_vector_type(4)));
typedef int          i32x8 __attribute__((ext_vector_type(8)));

#if defined(__has_builtin)
#  if __has_builtin(__builtin_amdgcn_tensor_load_to_lds) && __has_builtin(__builtin_amdgcn_s_wait_tensorcnt)
#    define HAVE_TDM 1
#  else
#    define HAVE_TDM 0
#  endif
#else
#  define HAVE_TDM 0
#endif

__device__ __forceinline__ __bf16 f2bf(float f) {
    unsigned u = __builtin_bit_cast(unsigned, f);
    unsigned r = u + 0x7FFFu + ((u >> 16) & 1u);   // round-to-nearest-even
    unsigned short h = (unsigned short)(r >> 16);
    return __builtin_bit_cast(__bf16, h);
}

// ---------------------------------------------------------------- fp32 -> bf16
__global__ void k_cvt_bf16(const float* __restrict__ in, __bf16* __restrict__ out, int n) {
    int i = blockIdx.x * blockDim.x + threadIdx.x;
    if (i < n) out[i] = f2bf(in[i]);
}

// ---------------------------------------------------------------- silu(c)
__global__ void k_silu(const float* __restrict__ in, float* __restrict__ out, int n) {
    int i = blockIdx.x * blockDim.x + threadIdx.x;
    if (i < n) { float v = in[i]; out[i] = v / (1.f + __expf(-v)); }
}

// ---------------------------------------------------------------- block reduction helper
__device__ __forceinline__ float blockSum(float v, float* red) {
    int tid = threadIdx.x;
    red[tid] = v; __syncthreads();
    for (int off = 128; off > 0; off >>= 1) {
        if (tid < off) red[tid] += red[tid + off];
        __syncthreads();
    }
    float r = red[0]; __syncthreads();
    return r;
}

// ---------------------------------------------------------------- ada = silu(c) @ Wada^T + bada  (block per output)
__global__ __launch_bounds__(256)
void k_ada(const float* __restrict__ sc, const float* __restrict__ Wada,
           const float* __restrict__ bada, float* __restrict__ ada) {
    __shared__ float red[256];
    int j = blockIdx.x;                    // 0 .. B*3D-1
    int b = j / (3 * D_);
    int o = j - b * (3 * D_);
    const float* w  = Wada + (size_t)o * (2 * D_);
    const float* cb = sc   + (size_t)b * (2 * D_);
    float acc = 0.f;
    for (int i = threadIdx.x; i < 2 * D_; i += 256) acc += cb[i] * w[i];
    float tot = blockSum(acc, red);
    if (threadIdx.x == 0) ada[j] = tot + bada[o];
}

// ---------------------------------------------------------------- fused LN1 -> modulate -> (skip) -> LN2 -> bf16
__global__ __launch_bounds__(256)
void k_ln(const float* __restrict__ x, const float* __restrict__ ada,
          const float* __restrict__ g1, const float* __restrict__ b1,
          const float* __restrict__ g2, const float* __restrict__ b2,
          float* __restrict__ skip, __bf16* __restrict__ xn) {
    __shared__ float red[256];
    int row = blockIdx.x;                      // 0 .. BS-1
    int b   = row >> 10;                       // S_ = 1024
    int tid = threadIdx.x;
    const float* xr    = x + (size_t)row * D_;
    const float* shift = ada + (size_t)b * 3 * D_;
    const float* scale = shift + D_;

    float v[4]; float s = 0.f, s2 = 0.f;
    #pragma unroll
    for (int i = 0; i < 4; ++i) {
        float t = xr[tid + i * 256];
        v[i] = t; s += t; s2 += t * t;
    }
    float mean = blockSum(s,  red) * (1.f / D_);
    float var  = blockSum(s2, red) * (1.f / D_) - mean * mean;
    float rstd = rsqrtf(var + 1e-5f);

    float w[4]; float t1 = 0.f, t2 = 0.f;
    #pragma unroll
    for (int i = 0; i < 4; ++i) {
        int col = tid + i * 256;
        float y = (v[i] - mean) * rstd * g1[col] + b1[col];
        y = y * (1.f + scale[col]) + shift[col];
        skip[(size_t)row * D_ + col] = y;
        w[i] = y; t1 += y; t2 += y * y;
    }
    float mean2 = blockSum(t1, red) * (1.f / D_);
    float var2  = blockSum(t2, red) * (1.f / D_) - mean2 * mean2;
    float rstd2 = rsqrtf(var2 + 1e-5f);
    #pragma unroll
    for (int i = 0; i < 4; ++i) {
        int col = tid + i * 256;
        float z = (w[i] - mean2) * rstd2 * g2[col] + b2[col];
        xn[(size_t)row * D_ + col] = f2bf(z);
    }
}

// ---------------------------------------------------------------- tiled WMMA bf16 GEMM
// C[M,Nout] = A[M,K](bf16) @ W[Nout,K]^T(bf16)  + epilogue
// block tile 128x64, 8 waves, 32x32 per wave (4 accumulators), ping-pong LDS,
// tiles staged by the Tensor Data Mover when available (ASYNC overlap with WMMA).
#define BM 128
#define BN 64
#define BK 32
#define LDT 40    // padded LDS row stride (halves) == TDM pad 4 DWORDs per 16 DWORDs

enum { EP_BF16_BIAS = 0, EP_SILU = 1, EP_BIAS_DUAL = 2, EP_SOFTPLUS = 3, EP_FINAL = 4 };

#if HAVE_TDM
// 2D tile load Global->LDS through the TDM (CDNA5 ISA).
// data_size=2B, pad_enable: 4 DWORDs pad per 16 DWORDs stored -> LDS stride 40 halves.
// tensor dims give hardware OOB-zero for ragged N tails.
__device__ __forceinline__ void tdm_load_2d(uint32_t lds_addr, const __bf16* gptr,
                                            int rem_k, int rem_rows,
                                            int tile_k, int tile_rows, int stride) {
    uint64_t ga = (uint64_t)(uintptr_t)gptr;
    u32x4 g0;
    g0.x = 0x1u;                                               // count=1, user mode
    g0.y = lds_addr;                                           // lds_addr[31:0]
    g0.z = (uint32_t)ga;                                       // global_addr[31:0]
    g0.w = ((uint32_t)(ga >> 32) & 0x01FFFFFFu) | (2u << 30);  // global_addr[56:32] | type=2
    i32x8 g1;
    g1[0] = (1 << 16)            // data_size = 2 bytes
          | (1 << 20)            // pad_enable
          | (3 << 22)            // pad_interval: 16 DWORDs
          | (3 << 25);           // pad_amount: 4 DWORDs
    g1[1] = (int)(((uint32_t)rem_k & 0xFFFFu) << 16);                               // tensor_dim0[15:0]
    g1[2] = (int)((((uint32_t)rem_k >> 16) & 0xFFFFu) |
                  (((uint32_t)rem_rows & 0xFFFFu) << 16));                          // dim0 hi | dim1 lo
    g1[3] = (int)((((uint32_t)rem_rows >> 16) & 0xFFFFu) |
                  ((uint32_t)tile_k << 16));                                        // dim1 hi | tile_dim0
    g1[4] = tile_rows;                                                              // tile_dim1 (tile_dim2=0)
    g1[5] = stride;                                                                 // tensor_dim0_stride[31:0]
    g1[6] = 0;
    g1[7] = 0;
    i32x4 z4 = {0, 0, 0, 0};
#if __clang_major__ >= 23
    i32x8 z8 = {0, 0, 0, 0, 0, 0, 0, 0};
    __builtin_amdgcn_tensor_load_to_lds(g0, g1, z4, z4, z8, 0);
#else
    __builtin_amdgcn_tensor_load_to_lds(g0, g1, z4, z4, 0);
#endif
}
#endif

__global__ __launch_bounds__(256)
void k_gemm(const __bf16* __restrict__ A, int lda,
            const __bf16* __restrict__ W, int K, int Nout,
            const float* __restrict__ bias, int mode,
            float* __restrict__ outF, __bf16* __restrict__ outB,
            const float* __restrict__ skip, const float* __restrict__ xin,
            const float* __restrict__ ada) {
    __shared__ __bf16 AsB[2][BM * LDT];
    __shared__ __bf16 BsB[2][BN * LDT];

    int tid  = threadIdx.x;
    int m0   = blockIdx.x * BM;
    int n0   = blockIdx.y * BN;
    int wave = tid >> 5;
    int lane = tid & 31;
    int wr   = (wave >> 1) * 32;          // wave row offset (4 groups x 32)
    int wc   = (wave & 1) * 32;           // wave col offset (2 groups x 32)
    int fm   = lane & 15;
    int fh   = lane >> 4;

    v8f acc[4];
    #pragma unroll
    for (int i = 0; i < 4; ++i) acc[i] = (v8f){};

    // ---- tile fill: TDM (one wave issues DMA) or per-thread load+ds_store ----
    auto fill = [&](int pg, int k0) {
#if HAVE_TDM
        if (tid < 32) {
            tdm_load_2d((uint32_t)(uintptr_t)&AsB[pg][0], A + (size_t)m0 * lda + k0,
                        K - k0, BS_ - m0, BK, BM, lda);
            tdm_load_2d((uint32_t)(uintptr_t)&BsB[pg][0], W + (size_t)n0 * K + k0,
                        K - k0, Nout - n0, BK, BN, K);
        }
#else
        {   // A tile: 128 rows x 32 halves, 32 bytes per thread
            int r  = tid >> 1;
            int ch = (tid & 1) * 16;
            const __bf16* pa = A + (size_t)(m0 + r) * lda + k0 + ch;
            uint4 va0 = *(const uint4*)pa;
            uint4 va1 = *(const uint4*)(pa + 8);
            if (k0 + BK < K) __builtin_prefetch(pa + BK, 0, 0);
            *(uint4*)(&AsB[pg][r * LDT + ch])     = va0;
            *(uint4*)(&AsB[pg][r * LDT + ch + 8]) = va1;
        }
        {   // B tile: 64 rows x 32 halves, 16 bytes per thread
            int r  = tid >> 2;
            int ch = (tid & 3) * 8;
            int wrow = n0 + r;
            uint4 vb = make_uint4(0, 0, 0, 0);
            if (wrow < Nout) {
                const __bf16* pb = W + (size_t)wrow * K + k0 + ch;
                vb = *(const uint4*)pb;
                if (k0 + BK < K) __builtin_prefetch(pb + BK, 0, 0);
            }
            *(uint4*)(&BsB[pg][r * LDT + ch]) = vb;
        }
#endif
    };
    auto sync_fill = [&]() {
#if HAVE_TDM
        if (tid < 32) __builtin_amdgcn_s_wait_tensorcnt(0);
#endif
        __syncthreads();
    };

    fill(0, 0);
    sync_fill();

    int pg = 0;
    for (int k0 = 0; k0 < K; k0 += BK) {
        if (k0 + BK < K) fill(pg ^ 1, k0 + BK);   // DMA next tile, overlaps WMMAs below

        const __bf16* As = &AsB[pg][0];
        const __bf16* Bs = &BsB[pg][0];
        // fragments per CDNA5 16-bit A/B VGPR layout (CDNA5 ISA §7.12.2)
        v16bf af[2], bfr[2];
        #pragma unroll
        for (int rj = 0; rj < 2; ++rj) {
            #pragma unroll
            for (int v = 0; v < 8; ++v) {
                int kk = ((v & 4) ? 16 : 0) + fh * 8 + (v & 3) * 2;
                af[rj][2 * v]      = As[(wr + rj * 16 + fm) * LDT + kk];
                af[rj][2 * v + 1]  = As[(wr + rj * 16 + fm) * LDT + kk + 1];
                bfr[rj][2 * v]     = Bs[(wc + rj * 16 + fm) * LDT + kk];
                bfr[rj][2 * v + 1] = Bs[(wc + rj * 16 + fm) * LDT + kk + 1];
            }
        }
        acc[0] = __builtin_amdgcn_wmma_f32_16x16x32_bf16(false, af[0], false, bfr[0], (short)0, acc[0], false, false);
        acc[1] = __builtin_amdgcn_wmma_f32_16x16x32_bf16(false, af[0], false, bfr[1], (short)0, acc[1], false, false);
        acc[2] = __builtin_amdgcn_wmma_f32_16x16x32_bf16(false, af[1], false, bfr[0], (short)0, acc[2], false, false);
        acc[3] = __builtin_amdgcn_wmma_f32_16x16x32_bf16(false, af[1], false, bfr[1], (short)0, acc[3], false, false);

        sync_fill();
        pg ^= 1;
    }

    float ac[32];
    #pragma unroll
    for (int i = 0; i < 4; ++i)
        #pragma unroll
        for (int v = 0; v < 8; ++v) ac[i * 8 + v] = acc[i][v];

    #pragma unroll
    for (int rj = 0; rj < 2; ++rj) {
        #pragma unroll
        for (int cj = 0; cj < 2; ++cj) {
            #pragma unroll
            for (int v = 0; v < 8; ++v) {
                int row = m0 + wr + rj * 16 + fh * 8 + v;
                int col = n0 + wc + cj * 16 + fm;
                if (col >= Nout) continue;
                float o = ac[(rj * 2 + cj) * 8 + v];
                if (bias) o += bias[col];
                size_t idx = (size_t)row * Nout + col;
                switch (mode) {
                    case EP_BF16_BIAS: outB[idx] = f2bf(o); break;
                    case EP_SILU:      outF[idx] = o / (1.f + __expf(-o)); break;
                    case EP_BIAS_DUAL: outF[idx] = o; outB[idx] = f2bf(o); break;
                    case EP_SOFTPLUS:  outF[idx] = (o > 20.f) ? o : log1pf(__expf(o)); break;
                    case EP_FINAL: {
                        float oo   = o + skip[idx];
                        int   b    = row >> 10;
                        float gate = ada[(size_t)b * 3 * D_ + 2 * D_ + col];
                        outF[idx]  = xin[idx] + gate * oo;
                    } break;
                }
            }
        }
    }
}

// ---------------------------------------------------------------- selective scan
// one channel (b,e) per 16-lane group, one state n per lane
__global__ __launch_bounds__(256)
void k_scan(const float* __restrict__ u, const float* __restrict__ delta,
            const float* __restrict__ dbc, const float* __restrict__ Alog,
            const float* __restrict__ Dp, float* __restrict__ prev,
            const float* __restrict__ actmx, __bf16* __restrict__ ycomb,
            int dir, int pass2) {
    int tid  = threadIdx.x;
    int chan = blockIdx.x * 16 + (tid >> 4);
    int n    = tid & 15;
    int b    = chan / E_;
    int e    = chan - b * E_;
    float Ac  = -__expf(Alog[e * N_ + n]);
    float dpe = Dp[e];
    float h = 0.f;
    for (int t = 0; t < S_; ++t) {
        int s = dir ? (S_ - 1 - t) : t;
        size_t ridx = (size_t)(b * S_ + s);
        float dlt = delta[ridx * E_ + e];
        float uu  = u[ridx * E_ + e];
        float Bn  = dbc[ridx * 96 + 64 + n];
        float Cn  = dbc[ridx * 96 + 80 + n];
        float dA  = __expf(dlt * Ac);
        h = dA * h + dlt * Bn * uu;
        float y = h * Cn;
        #pragma unroll
        for (int off = 8; off > 0; off >>= 1) y += __shfl_xor(y, off, 16);
        if (n == 0) {
            float  val = y + uu * dpe;
            size_t oi  = ridx * E_ + e;
            if (pass2) ycomb[oi] = f2bf(actmx[oi] * (prev[oi] + val));
            else       prev[oi]  = val;
        }
    }
}

// ---------------------------------------------------------------- host orchestration
extern "C" void kernel_launch(void* const* d_in, const int* in_sizes, int n_in,
                              void* d_out, int out_size, void* d_ws, size_t ws_size,
                              hipStream_t stream) {
    const float* x     = (const float*)d_in[0];
    const float* c     = (const float*)d_in[1];
    const float* n1g   = (const float*)d_in[3];
    const float* n1b   = (const float*)d_in[4];
    const float* n2g   = (const float*)d_in[5];
    const float* n2b   = (const float*)d_in[6];
    const float* Wx    = (const float*)d_in[7];
    const float* bx    = (const float*)d_in[8];
    const float* Wz    = (const float*)d_in[9];
    const float* bz    = (const float*)d_in[10];
    const float* Wf    = (const float*)d_in[11];
    const float* bff   = (const float*)d_in[12];
    const float* Wada  = (const float*)d_in[13];
    const float* bada  = (const float*)d_in[14];
    const float* Wc1   = (const float*)d_in[15];
    const float* bc1   = (const float*)d_in[16];
    const float* Wc2   = (const float*)d_in[17];
    const float* bc2   = (const float*)d_in[18];
    const float* Wdbc1 = (const float*)d_in[19];
    const float* Wdt1  = (const float*)d_in[20];
    const float* bdt1  = (const float*)d_in[21];
    const float* Alog1 = (const float*)d_in[22];
    const float* Dp1   = (const float*)d_in[23];
    const float* Wdbc2 = (const float*)d_in[24];
    const float* Wdt2  = (const float*)d_in[25];
    const float* bdt2  = (const float*)d_in[26];
    const float* Alog2 = (const float*)d_in[27];
    const float* Dp2   = (const float*)d_in[28];

    char* ws = (char*)d_ws;
    size_t off = 0;
    auto alloc = [&](size_t bytes) -> char* {
        char* p = ws + off;
        off = (off + bytes + 255) & ~(size_t)255;
        return p;
    };

    float*  sc     = (float*) alloc((size_t)B_ * 2 * D_ * 4);
    float*  adaB   = (float*) alloc((size_t)B_ * 3 * D_ * 4);
    float*  skip   = (float*) alloc((size_t)BS_ * D_ * 4);
    __bf16* xn     = (__bf16*)alloc((size_t)BS_ * D_ * 2);
    __bf16* mzb    = (__bf16*)alloc((size_t)BS_ * E_ * 2);
    float*  actmx  = (float*) alloc((size_t)BS_ * E_ * 4);
    float*  z1f    = (float*) alloc((size_t)BS_ * E_ * 4);
    __bf16* z1b    = (__bf16*)alloc((size_t)BS_ * E_ * 2);
    float*  z2f    = (float*) alloc((size_t)BS_ * E_ * 4);
    __bf16* z2b    = (__bf16*)alloc((size_t)BS_ * E_ * 2);
    float*  dbc1f  = (float*) alloc((size_t)BS_ * 96 * 4);
    __bf16* dbc1b  = (__bf16*)alloc((size_t)BS_ * 96 * 2);
    float*  dbc2f  = (float*) alloc((size_t)BS_ * 96 * 4);
    __bf16* dbc2b  = (__bf16*)alloc((size_t)BS_ * 96 * 2);
    float*  dl1    = (float*) alloc((size_t)BS_ * E_ * 4);
    float*  dl2    = (float*) alloc((size_t)BS_ * E_ * 4);
    float*  ssum   = (float*) alloc((size_t)BS_ * E_ * 4);
    __bf16* ycb    = (__bf16*)alloc((size_t)BS_ * E_ * 2);
    __bf16* wzb    = (__bf16*)alloc((size_t)E_ * D_ * 2);
    __bf16* wxb    = (__bf16*)alloc((size_t)E_ * D_ * 2);
    __bf16* wc1b   = (__bf16*)alloc((size_t)E_ * E_ * 2);
    __bf16* wc2b   = (__bf16*)alloc((size_t)E_ * E_ * 2);
    __bf16* wdbc1b = (__bf16*)alloc((size_t)96 * E_ * 2);
    __bf16* wdbc2b = (__bf16*)alloc((size_t)96 * E_ * 2);
    __bf16* wdt1b  = (__bf16*)alloc((size_t)E_ * R_ * 2);
    __bf16* wdt2b  = (__bf16*)alloc((size_t)E_ * R_ * 2);
    __bf16* wfb    = (__bf16*)alloc((size_t)D_ * E_ * 2);

    auto cvt = [&](const float* src, __bf16* dst, int n) {
        k_cvt_bf16<<<(n + 255) / 256, 256, 0, stream>>>(src, dst, n);
    };
    cvt(Wz,    wzb,    E_ * D_);
    cvt(Wx,    wxb,    E_ * D_);
    cvt(Wc1,   wc1b,   E_ * E_);
    cvt(Wc2,   wc2b,   E_ * E_);
    cvt(Wdbc1, wdbc1b, 96 * E_);
    cvt(Wdbc2, wdbc2b, 96 * E_);
    cvt(Wdt1,  wdt1b,  E_ * R_);
    cvt(Wdt2,  wdt2b,  E_ * R_);
    cvt(Wf,    wfb,    D_ * E_);

    k_silu<<<(B_ * 2 * D_ + 255) / 256, 256, 0, stream>>>(c, sc, B_ * 2 * D_);
    k_ada<<<B_ * 3 * D_, 256, 0, stream>>>(sc, Wada, bada, adaB);
    k_ln<<<BS_, 256, 0, stream>>>(x, adaB, n1g, n1b, n2g, n2b, skip, xn);

    dim3 blk(256);
    auto gemm = [&](const __bf16* A, int lda, const __bf16* W, int K, int Nout,
                    const float* bias, int mode, float* oF, __bf16* oB,
                    const float* sk, const float* xi, const float* ad) {
        dim3 grid(BS_ / BM, (Nout + BN - 1) / BN);
        k_gemm<<<grid, blk, 0, stream>>>(A, lda, W, K, Nout, bias, mode, oF, oB, sk, xi, ad);
    };

    gemm(xn,  D_, wzb,  D_, E_, bz,  EP_BF16_BIAS, nullptr, mzb,   nullptr, nullptr, nullptr);
    gemm(xn,  D_, wxb,  D_, E_, bx,  EP_SILU,      actmx,   nullptr, nullptr, nullptr, nullptr);
    gemm(mzb, E_, wc1b, E_, E_, bc1, EP_BIAS_DUAL, z1f, z1b, nullptr, nullptr, nullptr);
    gemm(mzb, E_, wc2b, E_, E_, bc2, EP_BIAS_DUAL, z2f, z2b, nullptr, nullptr, nullptr);
    gemm(z1b, E_, wdbc1b, E_, 96, nullptr, EP_BIAS_DUAL, dbc1f, dbc1b, nullptr, nullptr, nullptr);
    gemm(z2b, E_, wdbc2b, E_, 96, nullptr, EP_BIAS_DUAL, dbc2f, dbc2b, nullptr, nullptr, nullptr);
    gemm(dbc1b, 96, wdt1b, R_, E_, bdt1, EP_SOFTPLUS, dl1, nullptr, nullptr, nullptr, nullptr);
    gemm(dbc2b, 96, wdt2b, R_, E_, bdt2, EP_SOFTPLUS, dl2, nullptr, nullptr, nullptr, nullptr);

    k_scan<<<(B_ * E_) / 16, 256, 0, stream>>>(z1f, dl1, dbc1f, Alog1, Dp1, ssum, nullptr, nullptr, 0, 0);
    k_scan<<<(B_ * E_) / 16, 256, 0, stream>>>(z2f, dl2, dbc2f, Alog2, Dp2, ssum, actmx, ycb, 1, 1);

    gemm(ycb, E_, wfb, E_, D_, bff, EP_FINAL, (float*)d_out, nullptr, skip, x, adaB);

    (void)in_sizes; (void)n_in; (void)out_size; (void)ws_size;
}
